// NutritionalKnowledgeGraph_74320114090407
// MI455X (gfx1250) — matrix operand order repaired and checked by reading, not compiled, not run on hardware
//
#include <hip/hip_runtime.h>
#include <hip/hip_bf16.h>

typedef __attribute__((ext_vector_type(16))) __bf16 v16bf;
typedef __attribute__((ext_vector_type(8)))  float  v8f;

#define LN_EPS 1e-5f

// ---------------- degree / normalization ----------------

__global__ void k_init_deg(float* __restrict__ deg, int n) {
  int i = blockIdx.x * blockDim.x + threadIdx.x;
  if (i < n) deg[i] = 1.0f;   // self-loop contributes 1 to every degree
}

__global__ void k_deg_accum(const int* __restrict__ dst, float* __restrict__ deg, int E) {
  int e = blockIdx.x * blockDim.x + threadIdx.x;
  if (e < E) unsafeAtomicAdd(&deg[dst[e]], 1.0f);
}

__global__ void k_deg_finalize(float* __restrict__ deg, int n) {
  int i = blockIdx.x * blockDim.x + threadIdx.x;
  if (i < n) {
    float d = deg[i];
    deg[i] = (d > 0.0f) ? rsqrtf(d) : 0.0f;   // becomes dinv in place
  }
}

// ---------------- one-time converts --------------------------------------

// x0_bf16[n,f] = (bf16) emb[node_ids[n], f]
__global__ void k_gather_bf16(const float* __restrict__ emb, const int* __restrict__ ids,
                              __bf16* __restrict__ xbf, long long total, int F) {
  long long i = (long long)blockIdx.x * blockDim.x + threadIdx.x;
  if (i >= total) return;
  const int n = (int)(i / F);
  const int f = (int)(i % F);
  xbf[i] = (__bf16)emb[(size_t)ids[n] * F + f];
}

// Wt_bf16[n,k] = (bf16) W[k,n]   (so a B fragment is K-contiguous per lane)
__global__ void k_transpose_w_bf16(const float* __restrict__ W, __bf16* __restrict__ Wt,
                                   int K, int Ho) {
  int i = blockIdx.x * blockDim.x + threadIdx.x;
  if (i >= K * Ho) return;
  const int k = i / Ho;
  const int n = i % Ho;
  Wt[(size_t)n * K + k] = (__bf16)W[i];
}

// ---------------- WMMA GEMM: Hm[M,Ho] = X[M,K] @ W[K,Ho] ------------------
// X: bf16 row-major [M,K]; Wt: bf16 row-major [Ho,K] (= W transposed); Hm: f32.
// One wave owns Ho/128 16x16 C tiles (A fragment reused across them).

union AB32 { v16bf v; float4 f4[2]; };

template <int K, int Ho>
__global__ __launch_bounds__(256)
void k_gemm_wmma(const __bf16* __restrict__ X, const __bf16* __restrict__ Wt,
                 float* __restrict__ Hm, int M) {
  constexpr int NT = Ho / 128;                   // n-tiles per wave (1 or 2)
  const int lane  = threadIdx.x & 31;
  const int wave  = threadIdx.x >> 5;
  const int mtile = blockIdx.x;
  const int half  = lane >> 4;                   // 0: lanes 0-15, 1: lanes 16-31
  const int l15   = lane & 15;

  int m = mtile * 16 + l15;
  if (m >= M) m = M - 1;                         // clamp loads; stores guarded below
  // A fragment: lane holds row m, K = {kA..kA+7, kA+16..kA+23} per 32-chunk
  const __bf16* xrow = X + (size_t)m * K + (half ? 8 : 0);

  // B fragment: lane holds col ncol, K = {kB..kB+7, kB+16..kB+23} per 32-chunk
  const __bf16* wrow[NT];
#pragma unroll
  for (int j = 0; j < NT; ++j) {
    const int ncol = (wave + 8 * j) * 16 + l15;
    wrow[j] = Wt + (size_t)ncol * K + (half ? 8 : 0);
  }

  v8f acc[NT];
#pragma unroll
  for (int j = 0; j < NT; ++j) acc[j] = (v8f){};

#pragma unroll
  for (int k0 = 0; k0 < K; k0 += 32) {
    AB32 a;
    a.f4[0] = *reinterpret_cast<const float4*>(xrow + k0);        // K = kA..kA+7
    a.f4[1] = *reinterpret_cast<const float4*>(xrow + k0 + 16);   // K = kA+16..kA+23
#pragma unroll
    for (int j = 0; j < NT; ++j) {
      AB32 b;
      b.f4[0] = *reinterpret_cast<const float4*>(wrow[j] + k0);       // K = kB..kB+7
      b.f4[1] = *reinterpret_cast<const float4*>(wrow[j] + k0 + 16);  // K = kB+16..kB+23
      acc[j] = __builtin_amdgcn_wmma_f32_16x16x32_bf16(
          false, a.v, false, b.v, (short)0, acc[j], false, false);
    }
  }

  // C tile: element r of v8f -> row r0+r, col ncol (compile-time immediate offsets)
  const int r0 = mtile * 16 + (half ? 8 : 0);
  if (r0 + 8 <= M) {                             // uniform: whole tile in range
#pragma unroll
    for (int j = 0; j < NT; ++j) {
      float* o = Hm + (size_t)r0 * Ho + (wave + 8 * j) * 16 + l15;
#pragma unroll
      for (int r = 0; r < 8; ++r) o[r * Ho] = acc[j][r];
    }
  } else {
#pragma unroll
    for (int j = 0; j < NT; ++j) {
      float* o = Hm + (size_t)r0 * Ho + (wave + 8 * j) * 16 + l15;
#pragma unroll
      for (int r = 0; r < 8; ++r)
        if (r0 + r < M) o[r * Ho] = acc[j][r];
    }
  }
}

// ---------------- edge aggregation: agg[dst] += h[src]*dinv[src]*dinv[dst] ----

__global__ __launch_bounds__(256)
void k_aggregate(const int* __restrict__ edge, const float* __restrict__ dinv,
                 const float* __restrict__ Hm, float* __restrict__ agg,
                 int E, int Ho) {
  const int lane = threadIdx.x & 31;
  const int e    = blockIdx.x * 8 + (threadIdx.x >> 5);   // one wave per edge
  if (e >= E) return;
  const int   s = edge[e];
  const int   d = edge[E + e];
  const float w = dinv[s] * dinv[d];
  const float* hs = Hm  + (size_t)s * Ho;
  float*       ad = agg + (size_t)d * Ho;
  for (int f = lane * 4; f < Ho; f += 128) {              // coalesced float4 per lane
    const float4 v = *reinterpret_cast<const float4*>(hs + f);
    unsafeAtomicAdd(ad + f + 0, v.x * w);
    unsafeAtomicAdd(ad + f + 1, v.y * w);
    unsafeAtomicAdd(ad + f + 2, v.z * w);
    unsafeAtomicAdd(ad + f + 3, v.w * w);
  }
}

// ---------------- epilogue: x_bf16 = relu(agg + h*dinv^2 (self loop) + b) ----

__global__ void k_bias_relu_bf16(const float* __restrict__ agg, const float* __restrict__ Hm,
                                 const float* __restrict__ dinv, const float* __restrict__ b,
                                 __bf16* __restrict__ out, long long total, int Ho) {
  long long i = (long long)blockIdx.x * blockDim.x + threadIdx.x;
  if (i >= total) return;
  const int  row = (int)(i / Ho);
  const int  col = (int)(i % Ho);
  const float di = dinv[row];
  const float t  = agg[i] + Hm[i] * di * di + b[col];
  out[i] = (__bf16)(t > 0.0f ? t : 0.0f);
}

// ---------------- final epilogue: bias + self loop + LayerNorm (Ho == 128) ----

__global__ __launch_bounds__(256)
void k_bias_layernorm(const float* __restrict__ agg, const float* __restrict__ Hm,
                      const float* __restrict__ dinv, const float* __restrict__ b,
                      const float* __restrict__ gamma, const float* __restrict__ beta,
                      float* __restrict__ out, int Nn, int Ho) {
  const int lane = threadIdx.x & 31;
  const int row  = blockIdx.x * 8 + (threadIdx.x >> 5);   // one wave per row
  if (row >= Nn) return;
  const float di  = dinv[row];
  const float d2  = di * di;
  const float inv = 1.0f / (float)Ho;
  const size_t base = (size_t)row * Ho;
  const int f0 = lane * 4;                                 // 32 lanes * 4 = 128 = Ho

  float t[4];
#pragma unroll
  for (int j = 0; j < 4; ++j)
    t[j] = agg[base + f0 + j] + Hm[base + f0 + j] * d2 + b[f0 + j];

  float s = t[0] + t[1] + t[2] + t[3];
#pragma unroll
  for (int off = 16; off > 0; off >>= 1) s += __shfl_xor(s, off, 32);
  const float mu = s * inv;

  float q = 0.0f;
#pragma unroll
  for (int j = 0; j < 4; ++j) { const float dq = t[j] - mu; q += dq * dq; }
#pragma unroll
  for (int off = 16; off > 0; off >>= 1) q += __shfl_xor(q, off, 32);
  const float rstd = rsqrtf(q * inv + LN_EPS);

#pragma unroll
  for (int j = 0; j < 4; ++j)
    out[base + f0 + j] = (t[j] - mu) * rstd * gamma[f0 + j] + beta[f0 + j];
}

// ---------------- driver -----------------------------------------------------

extern "C" void kernel_launch(void* const* d_in, const int* in_sizes, int n_in,
                              void* d_out, int out_size, void* d_ws, size_t ws_size,
                              hipStream_t stream) {
  const int*   node_ids = (const int*)  d_in[0];
  const int*   edge     = (const int*)  d_in[1];   // [2,E]: src then dst
  const float* emb      = (const float*)d_in[2];
  const float* W1       = (const float*)d_in[3];
  const float* b1       = (const float*)d_in[4];
  const float* W2       = (const float*)d_in[5];
  const float* b2       = (const float*)d_in[6];
  const float* W3       = (const float*)d_in[7];
  const float* b3       = (const float*)d_in[8];
  const float* gamma    = (const float*)d_in[9];
  const float* beta     = (const float*)d_in[10];
  float* out = (float*)d_out;

  const int Nn = in_sizes[0];            // 50000
  const int E  = in_sizes[1] / 2;        // 800000
  const int F  = in_sizes[2] / Nn;       // 128 (fixed by reference)
  const int Hh = in_sizes[4];            // 256 (fixed by reference)
  const int Ho = in_sizes[8];            // 128 (fixed by reference)

  // workspace layout
  float*   ws     = (float*)d_ws;
  float*   dinv   = ws;                                  // Nn floats
  const size_t cap = (size_t)Nn * Hh;
  float*   hbuf   = ws + 65536;                          // Nn*Hh f32
  float*   aggbuf = hbuf + cap;                          // Nn*Hh f32
  __bf16*  xbf    = (__bf16*)(aggbuf + cap);             // Nn*Hh bf16
  __bf16*  wtbf   = xbf + cap;                           // Hh*Hh bf16 (transposed W)

  const dim3 blk(256);
  const int mtiles = (Nn + 15) / 16;
  const long long tH = (long long)Nn * Hh;

  // degree / symmetric normalization
  k_init_deg    <<<(Nn + 255) / 256, blk, 0, stream>>>(dinv, Nn);
  k_deg_accum   <<<(E  + 255) / 256, blk, 0, stream>>>(edge + E, dinv, E);
  k_deg_finalize<<<(Nn + 255) / 256, blk, 0, stream>>>(dinv, Nn);

  // ---- layer 1: h = emb[node_ids] @ W1 ; aggregate ; relu ----
  k_gather_bf16<<<(unsigned)(((long long)Nn * F + 255) / 256), blk, 0, stream>>>(
      emb, node_ids, xbf, (long long)Nn * F, F);
  k_transpose_w_bf16<<<(F * Hh + 255) / 256, blk, 0, stream>>>(W1, wtbf, F, Hh);
  k_gemm_wmma<128, 256><<<mtiles, blk, 0, stream>>>(xbf, wtbf, hbuf, Nn);
  hipMemsetAsync(aggbuf, 0, cap * sizeof(float), stream);
  k_aggregate<<<(E + 7) / 8, blk, 0, stream>>>(edge, dinv, hbuf, aggbuf, E, Hh);
  k_bias_relu_bf16<<<(unsigned)((tH + 255) / 256), blk, 0, stream>>>(
      aggbuf, hbuf, dinv, b1, xbf, tH, Hh);

  // ---- layer 2 ----
  k_transpose_w_bf16<<<(Hh * Hh + 255) / 256, blk, 0, stream>>>(W2, wtbf, Hh, Hh);
  k_gemm_wmma<256, 256><<<mtiles, blk, 0, stream>>>(xbf, wtbf, hbuf, Nn);
  hipMemsetAsync(aggbuf, 0, cap * sizeof(float), stream);
  k_aggregate<<<(E + 7) / 8, blk, 0, stream>>>(edge, dinv, hbuf, aggbuf, E, Hh);
  k_bias_relu_bf16<<<(unsigned)((tH + 255) / 256), blk, 0, stream>>>(
      aggbuf, hbuf, dinv, b2, xbf, tH, Hh);

  // ---- layer 3 + LayerNorm ----
  k_transpose_w_bf16<<<(Hh * Ho + 255) / 256, blk, 0, stream>>>(W3, wtbf, Hh, Ho);
  k_gemm_wmma<256, 128><<<mtiles, blk, 0, stream>>>(xbf, wtbf, hbuf, Nn);
  hipMemsetAsync(aggbuf, 0, (size_t)Nn * Ho * sizeof(float), stream);
  k_aggregate<<<(E + 7) / 8, blk, 0, stream>>>(edge, dinv, hbuf, aggbuf, E, Ho);
  k_bias_layernorm<<<(Nn + 7) / 8, blk, 0, stream>>>(
      aggbuf, hbuf, dinv, b3, gamma, beta, out, Nn, Ho);
}
